// EdgeEncoder_9964324127376
// MI455X (gfx1250) — matrix operand order repaired and checked by reading, not compiled, not run on hardware
//
#include <hip/hip_runtime.h>
#include <hip/hip_bf16.h>

typedef __attribute__((ext_vector_type(16))) _Float16 v16h;
typedef __attribute__((ext_vector_type(8)))  float    v8f;

#define N_NODES 16384
#define KNN     32
#define E_TOTAL (N_NODES * KNN)   // 524288
#define SLOPE   0.1f

// ---------------------------------------------------------------------------
// Kernel 1: convert W1 (256x128) and W2 (256x256) f32 -> f16 into workspace.
// Layout in ws: [W1h : 32768][W2h : 65536] halfs.
// ---------------------------------------------------------------------------
__global__ __launch_bounds__(256) void convert_w_kernel(const float* __restrict__ W1,
                                                        const float* __restrict__ W2,
                                                        _Float16* __restrict__ out) {
    int i = blockIdx.x * 256 + threadIdx.x;
    if (i < 32768) {
        out[i] = (_Float16)W1[i];
    } else if (i < 32768 + 65536) {
        out[i] = (_Float16)W2[i - 32768];
    }
}

// ---------------------------------------------------------------------------
// Kernel 2: brute-force kNN (k=32) over pos = geometric_info[:, 6:8].
// Each thread owns one query; points tiled through LDS (1024 pts = 8KB).
// Writes edge_index: idx_out[e] = source (neighbor), idx_out[E + e] = target.
// Insertion keeps ascending distance; strict compares + ascending j scan
// reproduce top_k tie-breaking (lower index first).
// ---------------------------------------------------------------------------
__global__ __launch_bounds__(256) void knn_kernel(const float* __restrict__ gi,
                                                  int* __restrict__ idx_out) {
    __shared__ float2 tile[1024];
    const int q = blockIdx.x * 256 + threadIdx.x;
    const float qx = gi[q * 10 + 6];
    const float qy = gi[q * 10 + 7];

    float bd[KNN];
    int   bi[KNN];
#pragma unroll
    for (int j = 0; j < KNN; ++j) { bd[j] = 3.0e38f; bi[j] = 0; }

    for (int t0 = 0; t0 < N_NODES; t0 += 1024) {
        __syncthreads();
        for (int j = threadIdx.x; j < 1024; j += 256) {
            const int g = t0 + j;
            tile[j] = make_float2(gi[g * 10 + 6], gi[g * 10 + 7]);
        }
        __syncthreads();
        for (int j = 0; j < 1024; ++j) {
            const int g = t0 + j;
            const float dx = tile[j].x - qx;
            const float dy = tile[j].y - qy;
            const float d = dx * dx + dy * dy;
            if (g != q && d < bd[KNN - 1]) {
                int p = KNN - 1;
                while (p > 0 && bd[p - 1] > d) {
                    bd[p] = bd[p - 1]; bi[p] = bi[p - 1]; --p;
                }
                bd[p] = d; bi[p] = g;
            }
        }
    }

    const long long ebase = (long long)q * KNN;
#pragma unroll
    for (int j = 0; j < KNN; ++j) {
        idx_out[ebase + j] = bi[j];                       // source
        idx_out[(long long)E_TOTAL + ebase + j] = q;      // target
    }
}

// ---------------------------------------------------------------------------
// WMMA fragment helpers (16x16x32 f16; A/B use this file's contiguous k order, not the ISA table order).
//   A (16x32, row M = l16):  elems i     -> K = 32*kt + 16*half + i (contig.)
//   B (32x16, col N = l16):  elems i     -> K = 32*kt + 16*half + i (contig.)
//   C/D (16x16 f32):         c[r] = C[M = r + 8*half][N = l16]
// ---------------------------------------------------------------------------
__device__ __forceinline__ v16h frag_A_lds(const _Float16* base, int stride,
                                           int kt, int l16, int half) {
    const _Float16* p = base + l16 * stride + kt * 32 + half * 16;
    union { uint4 u[2]; v16h h; } f;
    f.u[0] = *(const uint4*)(p);
    f.u[1] = *(const uint4*)(p + 8);
    return f.h;
}

__device__ __forceinline__ v16h frag_B_glb(const _Float16* __restrict__ W, int Kdim,
                                           int n0, int kt, int l16, int half) {
    const _Float16* p = W + (size_t)(n0 + l16) * Kdim + kt * 32 + half * 16;
    union { uint4 u[2]; v16h h; } f;
    f.u[0] = *(const uint4*)(p);
    f.u[1] = *(const uint4*)(p + 8);
    return f.h;
}

__device__ __forceinline__ float lrelu(float h) {
    return h >= 0.0f ? h : SLOPE * h;
}

// ---------------------------------------------------------------------------
// Kernel 3: fused edge-attr + 3-layer MLP.
// Block = 128 threads (4 waves of 32), 64 edges per block.
//   stage 0: raw 4-features for 64 edges (threads 0..63)   -> sraw
//   stage 1: layer0 (4->128) in f32 VALU, f16 out          -> sh0 (A layout)
//   stage 2: per-wave 16-edge tile: layer1 WMMA -> sh1, layer2 WMMA -> global
// ---------------------------------------------------------------------------
__global__ __launch_bounds__(128) void mlp_kernel(const float* __restrict__ gi,
                                                  const int* __restrict__ src_idx,
                                                  const float* __restrict__ W0,
                                                  const _Float16* __restrict__ wsh,
                                                  const float* __restrict__ b0,
                                                  const float* __restrict__ b1,
                                                  const float* __restrict__ b2,
                                                  float* __restrict__ out) {
    __shared__ __align__(16) float    sraw[64 * 4];     //  1 KB
    __shared__ __align__(16) _Float16 sh0[64 * 128];    // 16 KB
    __shared__ __align__(16) _Float16 sh1[64 * 256];    // 32 KB

    const _Float16* W1h = wsh;            // [256][128] f16
    const _Float16* W2h = wsh + 32768;    // [256][256] f16

    const int t = threadIdx.x;
    const long long eBase = (long long)blockIdx.x * 64;

    // ---- raw edge features ----
    if (t < 64) {
        const long long eg = eBase + t;
        const int s  = src_idx[eg];
        const int tg = (int)(eg >> 5);              // eg / k, k = 32
        const float* gs = gi + (size_t)s  * 10;
        const float* gt = gi + (size_t)tg * 10;
        sraw[t * 4 + 0] = (gs[6] - gt[6]) / gs[8];
        sraw[t * 4 + 1] = (gs[7] - gt[7]) / gs[9];
        sraw[t * 4 + 2] = logf(gs[4] / gt[4]);
        sraw[t * 4 + 3] = logf(gs[5] / gt[5]);
    }
    __syncthreads();

    // ---- layer 0: thread t owns output feature t (t in 0..127) ----
    {
        const float w0a = W0[t * 4 + 0];
        const float w0b = W0[t * 4 + 1];
        const float w0c = W0[t * 4 + 2];
        const float w0d = W0[t * 4 + 3];
        const float bb  = b0[t];
#pragma unroll 4
        for (int e = 0; e < 64; ++e) {
            float h = bb + sraw[e * 4 + 0] * w0a + sraw[e * 4 + 1] * w0b
                         + sraw[e * 4 + 2] * w0c + sraw[e * 4 + 3] * w0d;
            sh0[e * 128 + t] = (_Float16)lrelu(h);
        }
    }
    __syncthreads();

    // ---- per-wave WMMA tiles ----
    const int lane = t & 31;
    const int w    = t >> 5;          // wave id 0..3 -> edge tile
    const int half = lane >> 4;
    const int l16  = lane & 15;

    const _Float16* A0 = sh0 + w * 16 * 128;   // this wave's 16x128 h0 tile
    _Float16*       H1 = sh1 + w * 16 * 256;   // this wave's 16x256 h1 tile

    // layer 1: h1[16x256] = lrelu(h0[16x128] @ W1^T + b1)
    for (int nt = 0; nt < 16; ++nt) {
        v8f c = {};
#pragma unroll
        for (int kt = 0; kt < 4; ++kt) {
            v16h a = frag_A_lds(A0, 128, kt, l16, half);
            v16h b = frag_B_glb(W1h, 128, nt * 16, kt, l16, half);
            c = __builtin_amdgcn_wmma_f32_16x16x32_f16(false, a, false, b,
                                                       (short)0, c, false, false);
        }
        const float bias = b1[nt * 16 + l16];
#pragma unroll
        for (int r = 0; r < 8; ++r) {
            H1[(r + 8 * half) * 256 + nt * 16 + l16] = (_Float16)lrelu(c[r] + bias);
        }
    }
    // wave-private LDS slice: same-wave LDS ops are in-order, no barrier needed.

    // layer 2: out[16x256] = lrelu(h1[16x256] @ W2^T + b2)
    for (int nt = 0; nt < 16; ++nt) {
        v8f c = {};
#pragma unroll
        for (int kt = 0; kt < 8; ++kt) {
            v16h a = frag_A_lds(H1, 256, kt, l16, half);
            v16h b = frag_B_glb(W2h, 256, nt * 16, kt, l16, half);
            c = __builtin_amdgcn_wmma_f32_16x16x32_f16(false, a, false, b,
                                                       (short)0, c, false, false);
        }
        const float bias = b2[nt * 16 + l16];
#pragma unroll
        for (int r = 0; r < 8; ++r) {
            const long long edge = eBase + w * 16 + r + 8 * half;
            out[(size_t)edge * 256 + nt * 16 + l16] = lrelu(c[r] + bias);
        }
    }
}

// ---------------------------------------------------------------------------
// Launch. Inputs (setup_inputs order):
//   0:x (unused) 1:geometric_info 2:k 3:W0 4:b0 5:W1 6:b1 7:W2 8:b2
// d_out = [edge_attr: E*256 f32][edge_index: 2*E int32 (bit-stored)]
// ---------------------------------------------------------------------------
extern "C" void kernel_launch(void* const* d_in, const int* in_sizes, int n_in,
                              void* d_out, int out_size, void* d_ws, size_t ws_size,
                              hipStream_t stream) {
    const float* gi = (const float*)d_in[1];
    const float* W0 = (const float*)d_in[3];
    const float* b0 = (const float*)d_in[4];
    const float* W1 = (const float*)d_in[5];
    const float* b1 = (const float*)d_in[6];
    const float* W2 = (const float*)d_in[7];
    const float* b2 = (const float*)d_in[8];

    float*    attr = (float*)d_out;
    int*      eidx = (int*)d_out + (size_t)E_TOTAL * 256;
    _Float16* wsh  = (_Float16*)d_ws;   // 98304 halfs = 192 KB

    convert_w_kernel<<<(32768 + 65536 + 255) / 256, 256, 0, stream>>>(W1, W2, wsh);
    knn_kernel<<<N_NODES / 256, 256, 0, stream>>>(gi, eidx);
    mlp_kernel<<<E_TOTAL / 64, 128, 0, stream>>>(gi, eidx, W0, wsh, b0, b1, b2, attr);
}
